// CapsuleNet_34342558498919
// MI455X (gfx1250) — matrix operand order repaired and checked by reading, not compiled, not run on hardware
//
#include <hip/hip_runtime.h>

// ---------------------------------------------------------------------------
// CapsuleNet (conv + CBAM + primary caps + dynamic routing) for MI455X/gfx1250
// wave32; WMMA 16x16x32 f16 (conv-as-GEMM + routing reduction); TDM staging.
// ---------------------------------------------------------------------------

typedef _Float16 h16;
typedef __attribute__((ext_vector_type(16))) _Float16     v16h;
typedef __attribute__((ext_vector_type(8)))  _Float16     v8h;
typedef __attribute__((ext_vector_type(8)))  float        v8f;
typedef __attribute__((ext_vector_type(4)))  unsigned int u32x4;
typedef __attribute__((ext_vector_type(8)))  unsigned int u32x8;

#define BATCH 128
#define CCH   64
#define HH    134
#define WW    20
#define HWP   (HH * WW)           // 2680
#define NCAP  21440               // CCH*HWP/8
#define CHUNK_CAPS 4096           // caps per TDM tile (8 x 4096 f16 = 64 KB)

__device__ __forceinline__ v8f vz8() {
  v8f r;
#pragma unroll
  for (int i = 0; i < 8; ++i) r[i] = 0.f;
  return r;
}
__device__ __forceinline__ v16h vz16() {
  v16h r;
#pragma unroll
  for (int i = 0; i < 16; ++i) r[i] = (_Float16)0.f;
  return r;
}
__device__ __forceinline__ int imin(int a, int b) { return a < b ? a : b; }

// ------------------- TDM: 2-D tile load (groups 0+1 only) ------------------
// dsizeCode: 0=1B,1=2B,2=4B,3=8B. rowElems <= 65535 (tile_dim0 is 16-bit).
__device__ __forceinline__ void tdm_load_2d(unsigned lds_off,
                                            unsigned long long gaddr,
                                            unsigned rowElems, unsigned rows,
                                            unsigned long long strideElems,
                                            unsigned dsizeCode) {
  u32x4 g0;
  g0[0] = 1u;                                            // count=1, user desc
  g0[1] = lds_off;                                       // lds_addr
  g0[2] = (unsigned)(gaddr & 0xFFFFFFFFull);             // global_addr[31:0]
  g0[3] = (unsigned)((gaddr >> 32) & 0x1FFFFFFull) | (2u << 30); // [56:32],type=2
  u32x8 g1;
  g1[0] = (dsizeCode & 3u) << 16;                        // data_size
  g1[1] = (rowElems & 0xFFFFu) << 16;                    // tensor_dim0[15:0]
  g1[2] = ((rowElems >> 16) & 0xFFFFu) | ((rows & 0xFFFFu) << 16); // dim0 hi, dim1 lo
  g1[3] = ((rows >> 16) & 0xFFFFu) | ((rowElems & 0xFFFFu) << 16); // dim1 hi, tile_dim0
  g1[4] = rows & 0xFFFFu;                                // tile_dim1 (tile_dim2=0)
  g1[5] = (unsigned)(strideElems & 0xFFFFFFFFull);       // dim0_stride lo
  g1[6] = (unsigned)((strideElems >> 32) & 0xFFFFu);     // dim0_stride hi
  g1[7] = 0u;
  asm volatile("tensor_load_to_lds %0, %1" :: "s"(g0), "s"(g1) : "memory");
}

// ------------------------------ K0: zero stats -----------------------------
__global__ void k0_zero(float* __restrict__ a, int n) {
  int i = blockIdx.x * 256 + threadIdx.x;
  if (i < n) a[i] = 0.f;
}

// --------------- K1: 3x3 conv (1->64) as WMMA GEMM + ReLU + stats ----------
// A = weights [64 x 9]f16 (K padded to 32), B = im2col [9 x 16 px]f16.
// grid = (12, B); block = 256 (8 waves): wave w -> M-tile (w&3), 7 N-tiles.
__global__ __launch_bounds__(256) void k1_conv(const float* __restrict__ x,
                                               const float* __restrict__ cw,
                                               const float* __restrict__ cb,
                                               float* __restrict__ hgl,
                                               float* __restrict__ chansum,
                                               float* __restrict__ chanmax) {
  __shared__ float Xraw[HWP];            // TDM destination (f32)
  __shared__ h16   Xpad[137 * 22];       // zero-padded f16 image
  __shared__ h16   Wch[CCH * 9 + 16];    // +16 pad: unconditional gather reads
  __shared__ float Bias[CCH];
  const int b = blockIdx.y, tid = threadIdx.x;
  const int swid = __builtin_amdgcn_readfirstlane((int)tid) >> 5;

  if (swid == 0)   // DMA the whole image; overlaps with LDS init below
    tdm_load_2d((unsigned)(unsigned long long)(uintptr_t)Xraw,
                (unsigned long long)(uintptr_t)(x + (size_t)b * HWP),
                HWP, 1u, HWP, 2u);
  for (int i = tid; i < 137 * 22; i += 256) Xpad[i] = (h16)0.f;
  for (int i = tid; i < CCH * 9 + 16; i += 256)
    Wch[i] = (i < CCH * 9) ? (h16)cw[i] : (h16)0.f;
  if (tid < CCH) Bias[tid] = cb[tid];
  if (swid == 0) __builtin_amdgcn_s_wait_tensorcnt(0);
  __syncthreads();
  for (int i = tid; i < HWP; i += 256)
    Xpad[(i / WW + 1) * 22 + (i % WW) + 1] = (h16)Xraw[i];
  __syncthreads();

  const int wid = tid >> 5, lane = tid & 31;
  const int mi = wid & 3;                 // channel tile
  const int g = lane >> 4, m = lane & 15;
  const int chA = mi * 16 + m;            // A-fragment row

  // A fragment: elements 0..7 <-> K = 8g+e (valid iff < 9); 8..15 are K>=16 -> 0
  v16h af = vz16();
#pragma unroll
  for (int e = 0; e < 8; ++e) {
    h16 wv = Wch[chA * 9 + 8 * g + e];    // in-bounds thanks to pad
    af[e] = (8 * g + e < 9) ? wv : (h16)0.f;
  }
  float biasr[8];
#pragma unroll
  for (int i = 0; i < 8; ++i) biasr[i] = Bias[mi * 16 + i + 8 * g];

  float csum[8], cmax[8];
#pragma unroll
  for (int i = 0; i < 8; ++i) { csum[i] = 0.f; cmax[i] = 0.f; }

  const size_t chROW = ((size_t)(b * CCH + mi * 16 + 8 * g)) * HWP;
  for (int tt = (wid >> 2); tt < 14; tt += 2) {
    const int p0 = (blockIdx.x * 14 + tt) * 16;
    const int p = p0 + m;
    // B fragment: lanes g==0 hold K=0..7, lanes g==1 hold K=8 in element 0
    v16h bf = vz16();
    {
      const int py = p / WW, px = p % WW;
      const int xb = py * 22 + px;
      if (g == 0) {
#pragma unroll
        for (int kk = 0; kk < 8; ++kk)
          bf[kk] = Xpad[xb + (kk / 3) * 22 + (kk % 3)];
      } else {
        bf[0] = Xpad[xb + 2 * 22 + 2];
      }
    }
    v8f acc = vz8();
    acc = __builtin_amdgcn_wmma_f32_16x16x32_f16(false, af, false, bf,
                                                 (short)0, acc, false, false);
    // single EXEC-masked region for all 8 stores (p doesn't depend on i);
    // all-true for the 167 full tiles, partial only on the last tile
    if (p < HWP) {
#pragma unroll
      for (int i = 0; i < 8; ++i) {
        float val = fmaxf(acc[i] + biasr[i], 0.f);
        hgl[chROW + (size_t)i * HWP + p] = val;
        csum[i] += val;
        cmax[i] = fmaxf(cmax[i], val);
      }
    }
  }
  // lanes 0..15 of each half share the channel: shfl-reduce then one atomic
#pragma unroll
  for (int i = 0; i < 8; ++i) {
    float s = csum[i], mx = cmax[i];
#pragma unroll
    for (int mk = 1; mk < 16; mk <<= 1) {
      s += __shfl_xor(s, mk, 32);
      mx = fmaxf(mx, __shfl_xor(mx, mk, 32));
    }
    if (m == 0) {
      int ch = mi * 16 + i + 8 * g;
      atomicAdd(&chansum[b * CCH + ch], s);
      atomicMax((int*)&chanmax[b * CCH + ch], __float_as_int(mx)); // vals >= 0
    }
  }
}

// ------------------- K2: channel-attention MLP + sigmoid -------------------
__global__ void k2_ca(const float* __restrict__ chansum,
                      const float* __restrict__ chanmax,
                      const float* __restrict__ w1, const float* __restrict__ w2,
                      float* __restrict__ ca) {
  __shared__ float avg[CCH], mxv[CCH], ha[4], hm[4];
  const int b = blockIdx.x, t = threadIdx.x;   // 64 threads
  avg[t] = chansum[b * CCH + t] * (1.f / HWP);
  mxv[t] = chanmax[b * CCH + t];
  __syncthreads();
  if (t < 4) {
    float sa2 = 0.f, sm = 0.f;
    for (int c = 0; c < CCH; ++c) {
      sa2 += avg[c] * w1[t * CCH + c];
      sm  += mxv[c] * w1[t * CCH + c];
    }
    ha[t] = fmaxf(sa2, 0.f);
    hm[t] = fmaxf(sm, 0.f);
  }
  __syncthreads();
  float oa = 0.f, om = 0.f;
  for (int k = 0; k < 4; ++k) { oa += ha[k] * w2[t * 4 + k]; om += hm[k] * w2[t * 4 + k]; }
  ca[b * CCH + t] = 1.f / (1.f + expf(-(oa + om)));
}

// --------- K3: apply channel attention in-place + spatial mean/max ---------
__global__ __launch_bounds__(256) void k3_spstats(float* __restrict__ hgl,
                                                  const float* __restrict__ ca,
                                                  float* __restrict__ sp) {
  const int b = blockIdx.y;
  const int p = blockIdx.x * 256 + threadIdx.x;
  if (p >= HWP) return;
  float s = 0.f, mx = -1e30f;
  for (int c = 0; c < CCH; ++c) {
    size_t idx = ((size_t)(b * CCH + c)) * HWP + p;
    float v = hgl[idx] * ca[b * CCH + c];
    hgl[idx] = v;
    s += v;
    mx = fmaxf(mx, v);
  }
  sp[((size_t)b * 2 + 0) * HWP + p] = s * (1.f / CCH);
  sp[((size_t)b * 2 + 1) * HWP + p] = mx;
}

// ----------------- K4: 7x7 spatial-attention conv + sigmoid ----------------
// zero-padded LDS (2 x 140 x 26) -> all 98 taps branch-free
__global__ __launch_bounds__(256) void k4_saconv(const float* __restrict__ sp,
                                                 const float* __restrict__ saw,
                                                 float* __restrict__ sa) {
  __shared__ float spp[2 * 140 * 26];   // 29.1 KB, zero-padded by 3 each side
  __shared__ float wsa[98];
  const int b = blockIdx.x, tid = threadIdx.x;
  for (int i = tid; i < 2 * 140 * 26; i += 256) spp[i] = 0.f;
  if (tid < 98) wsa[tid] = saw[tid];
  __syncthreads();
  for (int i = tid; i < 2 * HWP; i += 256) {
    int ic = i / HWP, p = i % HWP;
    spp[ic * (140 * 26) + (p / WW + 3) * 26 + (p % WW) + 3] =
        sp[(size_t)b * 2 * HWP + i];
  }
  __syncthreads();
  for (int p = tid; p < HWP; p += 256) {
    const int y = p / WW, x = p % WW;
    float acc = 0.f;
#pragma unroll
    for (int ic = 0; ic < 2; ++ic) {
      const int base = ic * (140 * 26) + y * 26 + x;
#pragma unroll
      for (int dy = 0; dy < 7; ++dy)
#pragma unroll
        for (int dx = 0; dx < 7; ++dx)
          acc += spp[base + dy * 26 + dx] * wsa[ic * 49 + dy * 7 + dx];
    }
    sa[(size_t)b * HWP + p] = 1.f / (1.f + expf(-acc));
  }
}

// ------ K5: apply spatial attention, emit u in k-major f16 (LDS tiled) -----
// output layout: u16T[(b*8 + k)*NCAP + n],  n = capsule, k = 0..7
__global__ __launch_bounds__(256) void k5_u16(const float* __restrict__ hgl,
                                              const float* __restrict__ sa,
                                              h16* __restrict__ u16T) {
  __shared__ h16 tile[256 * 9];              // stride 9 to dodge bank conflicts
  const int tid = threadIdx.x;
  const size_t base = (size_t)blockIdx.x * 2048;
  const size_t sampStride = (size_t)CCH * HWP;
#pragma unroll
  for (int r = 0; r < 8; ++r) {
    size_t idx = base + (size_t)r * 256 + tid;
    size_t bs = idx / sampStride;
    int p = (int)((idx - bs * sampStride) % HWP);
    float v = hgl[idx] * sa[bs * HWP + p];
    int local = r * 256 + tid;
    tile[(local >> 3) * 9 + (local & 7)] = (h16)v;
  }
  __syncthreads();
  const size_t cap = (base >> 3) + tid;
  const size_t bs = cap / NCAP;
  const size_t n = cap - bs * NCAP;
#pragma unroll
  for (int k = 0; k < 8; ++k)
    u16T[(bs * 8 + k) * NCAP + n] = tile[tid * 9 + k];
}

// -------- K6: dynamic routing, fused per sample; u_hat never materialized --
// A rows 0/1 = coupling coeffs (f16 LDS rows), B = 2-D TDM-staged u tile.
__device__ __forceinline__ v8f route_step(const h16* carr0, const h16* carr1,
                                          const h16* ustT, int gbase, int n0,
                                          int validCaps, int g, int m, v8f acc) {
  v16h af = vz16();
  if (m < 2) {                       // rows 0/1 only: two 16-byte LDS loads
    const h16* rowp = m ? carr1 : carr0;
    v8h lo = *(const v8h*)(rowp + gbase + n0 + 16 * g);
    v8h hi = *(const v8h*)(rowp + gbase + n0 + 16 * g + 8);
#pragma unroll
    for (int e = 0; e < 8; ++e) { af[e] = lo[e]; af[8 + e] = hi[e]; }
  }
  // B fragment: one aligned 32-byte LDS load; lanes m>=8 duplicate rows m&7
  // (their C columns are never read, so no masking needed)
  v16h bf = *(const v16h*)(ustT + (size_t)(m & 7) * validCaps + n0 + 16 * g);
  return __builtin_amdgcn_wmma_f32_16x16x32_f16(false, af, false, bf,
                                                (short)0, acc, false, false);
}

__global__ __launch_bounds__(256) void k6_route(const h16* __restrict__ u16T,
                                                const float* __restrict__ capsW,
                                                float* __restrict__ out) {
  extern __shared__ char smem[];
  float* diff  = (float*)smem;                    // NCAP f32          (85.8 KB)
  h16*   carr0 = (h16*)(diff + NCAP);             // NCAP f16: c0      (42.9 KB)
  h16*   carr1 = carr0 + NCAP;                    // NCAP f16: c1      (42.9 KB)
  h16*   ustT  = carr1 + NCAP;                    // 8 x CHUNK_CAPS    (64 KB)
  float* Wc    = (float*)(ustT + 8 * CHUNK_CAPS); // caps_W [8][32]
  float* t_sm  = Wc + 256;                        // 16
  float* v_sm  = t_sm + 16;                       // 32
  float* gd_sm = v_sm + 32;                       // 8

  const int b = blockIdx.x, tid = threadIdx.x;
  const int wid = tid >> 5, lane = tid & 31;
  const int g = lane >> 4, m = lane & 15;
  for (int i = tid; i < NCAP; i += 256) {
    diff[i] = 0.f;
    carr0[i] = (h16)0.5f;               // softmax of zero logits
    carr1[i] = (h16)0.5f;
  }
  for (int i = tid; i < 256; i += 256) Wc[i] = capsW[i];
  __syncthreads();

  const h16* ubT = u16T + (size_t)b * 8 * NCAP;
  const unsigned ustOff = (unsigned)(unsigned long long)(uintptr_t)ustT;
  const int swid = __builtin_amdgcn_readfirstlane((int)tid) >> 5;

  for (int it = 0; it < 3; ++it) {
    if (tid < 16) t_sm[tid] = 0.f;
    __syncthreads();

    v8f acc0 = vz8(), acc1 = vz8();
    const int nchunks = (NCAP + CHUNK_CAPS - 1) / CHUNK_CAPS;   // 6
    for (int ck = 0; ck < nchunks; ++ck) {
      const int base = ck * CHUNK_CAPS;
      const int validCaps = imin(CHUNK_CAPS, NCAP - base);      // 4096 or 960
      if (swid == 0) {    // TDM ignores EXEC -> scalar branch, wave 0 only
        tdm_load_2d(ustOff, (unsigned long long)(uintptr_t)(ubT + base),
                    (unsigned)validCaps, 8u, (unsigned long long)NCAP, 1u);
        __builtin_amdgcn_s_wait_tensorcnt(0);
      }
      __syncthreads();
      // validCaps is a multiple of 32 -> no per-element guards anywhere
      for (int s = wid; 32 * s < validCaps; s += 16) {
        acc0 = route_step(carr0, carr1, ustT, base, 32 * s, validCaps, g, m, acc0);
        if (32 * (s + 8) < validCaps)
          acc1 = route_step(carr0, carr1, ustT, base, 32 * (s + 8), validCaps, g, m, acc1);
      }
      __syncthreads();    // all waves done reading ustT before next TDM
    }
    // C layout: VGPR i, lane l -> (M = i + 8*(l/16), N = l%16); t at M=0,1 N<8
    if (lane < 8) {
      atomicAdd(&t_sm[lane],     acc0[0] + acc1[0]);
      atomicAdd(&t_sm[8 + lane], acc0[1] + acc1[1]);
    }
    __syncthreads();

    // s_jd = sum_k t_jk W[k, j*16+d]; squash; (final iter) capsule lengths
    if (tid < 32) {
      const int j = tid >> 4, d = tid & 15;
      float s = 0.f;
#pragma unroll
      for (int k = 0; k < 8; ++k) s += t_sm[j * 8 + k] * Wc[k * 32 + j * 16 + d];
      float ss = s * s;
#pragma unroll
      for (int mk = 1; mk < 16; mk <<= 1) ss += __shfl_xor(ss, mk, 32);
      ss += 1e-8f;
      const float scale = sqrtf(ss) / (1.f + ss);
      const float vv = scale * s;
      v_sm[tid] = vv;
      if (it == 2) {
        float vs = vv * vv;
#pragma unroll
        for (int mk = 1; mk < 16; mk <<= 1) vs += __shfl_xor(vs, mk, 32);
        if (d == 0) out[b * 2 + j] = sqrtf(vs + 1e-8f);
      }
    }
    __syncthreads();

    if (it < 2) {
      if (tid < 8) {   // gd_k = g1k - g0k,  g_jk = sum_d v_jd W[k, j*16+d]
        float g0 = 0.f, g1 = 0.f;
#pragma unroll
        for (int d2 = 0; d2 < 16; ++d2) {
          g0 += v_sm[d2]      * Wc[tid * 32 + d2];
          g1 += v_sm[16 + d2] * Wc[tid * 32 + 16 + d2];
        }
        gd_sm[tid] = g1 - g0;
      }
      __syncthreads();
      // fused: diff update + next iteration's coupling coefficients
      for (int n = tid; n < NCAP; n += 256) {
        float dot = 0.f;
#pragma unroll
        for (int k = 0; k < 8; ++k) dot += (float)ubT[k * NCAP + n] * gd_sm[k];
        float d = diff[n] + dot;
        diff[n] = d;
        float c1 = 1.f / (1.f + __expf(-d));
        carr1[n] = (h16)c1;
        carr0[n] = (h16)(1.f - c1);
      }
      __syncthreads();
    }
  }
}

// ------------------------------- launcher ----------------------------------
extern "C" void kernel_launch(void* const* d_in, const int* in_sizes, int n_in,
                              void* d_out, int out_size, void* d_ws, size_t ws_size,
                              hipStream_t stream) {
  (void)in_sizes; (void)n_in; (void)out_size; (void)ws_size;
  const float* x      = (const float*)d_in[0];
  const float* conv_w = (const float*)d_in[1];
  const float* conv_b = (const float*)d_in[2];
  const float* ca_w1  = (const float*)d_in[3];
  const float* ca_w2  = (const float*)d_in[4];
  const float* sa_w   = (const float*)d_in[5];
  const float* caps_W = (const float*)d_in[6];
  float* out = (float*)d_out;

  float* ws      = (float*)d_ws;
  float* hbuf    = ws;                                        // 21,954,560 f32
  float* chansum = hbuf + (size_t)BATCH * CCH * HWP;          // 8192
  float* chanmax = chansum + BATCH * CCH;                     // 8192
  float* cab     = chanmax + BATCH * CCH;                     // 8192
  float* spb     = cab + BATCH * CCH;                         // 686,080
  float* sab     = spb + (size_t)BATCH * 2 * HWP;             // 343,040
  h16*   u16T    = (h16*)(sab + (size_t)BATCH * HWP);         // 21,954,560 f16

  k0_zero<<<(2 * BATCH * CCH + 255) / 256, 256, 0, stream>>>(chansum, 2 * BATCH * CCH);
  k1_conv<<<dim3(12, BATCH), 256, 0, stream>>>(x, conv_w, conv_b, hbuf, chansum, chanmax);
  k2_ca<<<BATCH, 64, 0, stream>>>(chansum, chanmax, ca_w1, ca_w2, cab);
  k3_spstats<<<dim3((HWP + 255) / 256, BATCH), 256, 0, stream>>>(hbuf, cab, spb);
  k4_saconv<<<BATCH, 256, 0, stream>>>(spb, sa_w, sab);
  k5_u16<<<10720, 256, 0, stream>>>(hbuf, sab, u16T);  // 21,954,560 / 2048

  size_t smem6 = (size_t)NCAP * 4 + (size_t)NCAP * 2 * 2 +
                 (size_t)CHUNK_CAPS * 8 * 2 + (256 + 16 + 32 + 8) * 4; // 238,304 B
  k6_route<<<BATCH, 256, smem6, stream>>>(u16T, caps_W, out);
}
